// FeedForward_75763223101598
// MI455X (gfx1250) — matrix operand order repaired and checked by reading, not compiled, not run on hardware
//
#include <hip/hip_runtime.h>
#include <hip/hip_bf16.h>

// out = rownorm( relu(x @ w1^T + b1) + x ),  x:(16384,4096) w1:(4096,4096) b1:(4096)
// Compute-bound GEMM via bf16 hi/lo split-precision WMMA (3 wmma per tile pair
// -> ~fp32 accuracy at 3/8 the instruction cost of native f32 WMMA).
// 256x128 workgroup tile, 8 waves of 64x64, double-buffered LDS (96KB dynamic),
// one barrier per K-stage with ds_store overlapped with WMMA.

typedef __attribute__((ext_vector_type(16))) __bf16 v16bf;
typedef __attribute__((ext_vector_type(8)))  float  v8f;

#define NROW 16384
#define FDIM 4096
#define MB   256
#define NB   128
#define KB   32
#define S_ITERS (FDIM / KB)     // 128 K-stages

// LDS layout (ushort units), per buffer: A_hi[256*32] A_lo[256*32] B_hi[128*32] B_lo[128*32]
#define A_HI 0
#define A_LO (MB * KB)          // 8192
#define B_HI (2 * MB * KB)      // 16384
#define B_LO (2 * MB * KB + NB * KB)
#define BUF_STRIDE (2 * MB * KB + 2 * NB * KB)   // 24576 ushorts = 48KB
#define LDS_BYTES (2 * BUF_STRIDE * 2)           // 96KB

union FragBF {
  uint4 q[2];
  v16bf v;
};

// granule-swizzled LDS offset: 8-bf16 (16B) granules XORed by row bits [3:2]
__device__ __forceinline__ int ldsoff(int row, int g) {
  return row * KB + ((g ^ ((row >> 2) & 3)) << 3);
}

// Split fp32 pair into packed bf16-hi dword and bf16-lo dword.
// hi = trunc-bf16(x); lo = trunc-bf16(x - hi); (x - hi) is exact in fp32.
__device__ __forceinline__ void split_pack(float a, float b, unsigned &h, unsigned &l) {
  unsigned ua = __float_as_uint(a), ub = __float_as_uint(b);
  h = __builtin_amdgcn_perm(ub, ua, 0x07060302u);       // {ub[31:16], ua[31:16]}
  float ra = a - __uint_as_float(ua & 0xFFFF0000u);
  float rb = b - __uint_as_float(ub & 0xFFFF0000u);
  l = __builtin_amdgcn_perm(__float_as_uint(rb), __float_as_uint(ra), 0x07060302u);
}

__global__ __launch_bounds__(256, 1)
void ffn_gemm_kernel(const float* __restrict__ x, const float* __restrict__ w1,
                     const float* __restrict__ b1, float* __restrict__ out)
{
  extern __shared__ __align__(16) unsigned short smem[];

  const int t    = threadIdx.x;
  const int lane = t & 31;
  const int wave = t >> 5;
  const int wm   = wave >> 1;        // 0..3 : 64-row slab of the 256-row tile
  const int wn   = wave & 1;         // 0..1 : 64-col slab of the 128-col tile
  const int m0   = blockIdx.y * MB;
  const int e0   = blockIdx.x * NB;

  // ---- staging: per stage each thread stages 32 A floats (two half-rows) + 16 B floats
  const int srow = t >> 1;           // 0..127
  const int skh  = (t & 1) * 16;     // K half: 0 or 16
  const int sg0  = skh >> 3;         // granule 0 or 2
  const float* gA0 = x  + (size_t)(m0 + srow)       * FDIM + skh;
  const float* gA1 = x  + (size_t)(m0 + srow + 128) * FDIM + skh;
  const float* gB  = w1 + (size_t)(e0 + srow)       * FDIM + skh;

  const int fr   = lane & 15;        // row/col within a 16-wide fragment
  const int half = lane >> 4;        // 0 or 1

  v8f acc[4][4];
  #pragma unroll
  for (int i = 0; i < 4; ++i)
    #pragma unroll
    for (int j = 0; j < 4; ++j)
      #pragma unroll
      for (int k = 0; k < 8; ++k) acc[i][j][k] = 0.0f;

  float4 a0[4], a1[4], bb[4];        // staging registers (48 fp32)

  auto load_stage = [&](int s) {
    const float4* pa0 = (const float4*)(gA0 + s * KB);
    const float4* pa1 = (const float4*)(gA1 + s * KB);
    const float4* pb  = (const float4*)(gB  + s * KB);
    #pragma unroll
    for (int i = 0; i < 4; ++i) { a0[i] = pa0[i]; a1[i] = pa1[i]; bb[i] = pb[i]; }
  };

  auto stage16 = [&](unsigned short* base_hi, unsigned short* base_lo,
                     const float4* f, int row) {
    #pragma unroll
    for (int q = 0; q < 2; ++q) {
      uint4 hi, lo;
      float4 f0 = f[2 * q], f1 = f[2 * q + 1];
      split_pack(f0.x, f0.y, hi.x, lo.x);
      split_pack(f0.z, f0.w, hi.y, lo.y);
      split_pack(f1.x, f1.y, hi.z, lo.z);
      split_pack(f1.z, f1.w, hi.w, lo.w);
      const int off = ldsoff(row, sg0 + q);
      *(uint4*)&base_hi[off] = hi;
      *(uint4*)&base_lo[off] = lo;
    }
  };

  auto store_stage = [&](int buf) {
    unsigned short* b = smem + buf * BUF_STRIDE;
    stage16(b + A_HI, b + A_LO, a0, srow);
    stage16(b + A_HI, b + A_LO, a1, srow + 128);
    stage16(b + B_HI, b + B_LO, bb, srow);
  };

  auto compute_stage = [&](int buf) {
    const unsigned short* bs = smem + buf * BUF_STRIDE;
    FragBF Ah[4], Al[4];
    // A fragments (16x32 bf16): lane<16 -> K 0..7 & 16..23 ; lane>=16 -> K 8..15 & 24..31
    #pragma unroll
    for (int tm = 0; tm < 4; ++tm) {
      const int row = wm * 64 + tm * 16 + fr;
      const int o0  = ldsoff(row, half);
      const int o1  = ldsoff(row, half + 2);
      Ah[tm].q[0] = *(const uint4*)&bs[A_HI + o0];
      Ah[tm].q[1] = *(const uint4*)&bs[A_HI + o1];
      Al[tm].q[0] = *(const uint4*)&bs[A_LO + o0];
      Al[tm].q[1] = *(const uint4*)&bs[A_LO + o1];
    }
    // B fragments (32x16 bf16): lane<16 -> K 0..7 & 16..23 ; lane>=16 -> K 8..15 & 24..31
    #pragma unroll
    for (int tn = 0; tn < 4; ++tn) {
      const int col = wn * 64 + tn * 16 + fr;
      const int o0  = ldsoff(col, half);
      const int o1  = ldsoff(col, half + 2);
      FragBF Bh, Bl;
      Bh.q[0] = *(const uint4*)&bs[B_HI + o0];
      Bh.q[1] = *(const uint4*)&bs[B_HI + o1];
      Bl.q[0] = *(const uint4*)&bs[B_LO + o0];
      Bl.q[1] = *(const uint4*)&bs[B_LO + o1];
      #pragma unroll
      for (int tm = 0; tm < 4; ++tm) {
        v8f c = acc[tm][tn];
        c = __builtin_amdgcn_wmma_f32_16x16x32_bf16(false, Ah[tm].v, false, Bh.v,
                                                    (short)0, c, false, false);
        c = __builtin_amdgcn_wmma_f32_16x16x32_bf16(false, Ah[tm].v, false, Bl.v,
                                                    (short)0, c, false, false);
        c = __builtin_amdgcn_wmma_f32_16x16x32_bf16(false, Al[tm].v, false, Bh.v,
                                                    (short)0, c, false, false);
        acc[tm][tn] = c;
      }
    }
  };

  // ---- software pipeline: one barrier per stage; ds_store overlaps WMMA
  load_stage(0);
  store_stage(0);
  load_stage(1);
  __syncthreads();
  int p = 0;
  for (int s = 0; s < S_ITERS; ++s) {
    if (s + 1 < S_ITERS) store_stage(p ^ 1);   // stage s+1 (regs from last iter)
    if (s + 2 < S_ITERS) load_stage(s + 2);    // prefetch stage s+2 into regs
    compute_stage(p);                          // 48 WMMAs from buffer p
    __syncthreads();
    p ^= 1;
  }

  // ---- fused epilogue: + bias, relu, + residual x, store pre-norm residual
  #pragma unroll
  for (int tn = 0; tn < 4; ++tn) {
    const int e = e0 + wn * 64 + tn * 16 + fr;
    const float bias = b1[e];
    #pragma unroll
    for (int tm = 0; tm < 4; ++tm) {
      const int mb = m0 + wm * 64 + tm * 16 + half * 8;
      #pragma unroll
      for (int v = 0; v < 8; ++v) {
        const size_t idx = (size_t)(mb + v) * FDIM + e;
        float val = acc[tm][tn][v] + bias;
        val = fmaxf(val, 0.0f) + x[idx];
        out[idx] = val;
      }
    }
  }
}

// one block per row: compute mean / E[x^2], normalize in place
__global__ __launch_bounds__(256)
void rownorm_kernel(float* __restrict__ out)
{
  __shared__ float red_s[8];
  __shared__ float red_q[8];
  const int row = blockIdx.x;
  const int t = threadIdx.x;
  float* p = out + (size_t)row * FDIM + t * 16;

  float4 v[4];
  #pragma unroll
  for (int i = 0; i < 4; ++i) v[i] = ((float4*)p)[i];

  float s = 0.f, q = 0.f;
  #pragma unroll
  for (int i = 0; i < 4; ++i) {
    s += v[i].x + v[i].y + v[i].z + v[i].w;
    q += v[i].x * v[i].x + v[i].y * v[i].y + v[i].z * v[i].z + v[i].w * v[i].w;
  }
  #pragma unroll
  for (int off = 16; off >= 1; off >>= 1) {
    s += __shfl_xor(s, off, 32);
    q += __shfl_xor(q, off, 32);
  }
  const int wave = t >> 5, lane = t & 31;
  if (lane == 0) { red_s[wave] = s; red_q[wave] = q; }
  __syncthreads();
  s = 0.f; q = 0.f;
  #pragma unroll
  for (int i = 0; i < 8; ++i) { s += red_s[i]; q += red_q[i]; }

  const float mean = s * (1.0f / FDIM);
  const float var  = q * (1.0f / FDIM) - mean * mean;   // ddof=0, matches np.std
  const float inv  = rsqrtf(var + 1e-4f);               // sqrt(std^2 + EPS)

  #pragma unroll
  for (int i = 0; i < 4; ++i) {
    v[i].x = (v[i].x - mean) * inv;
    v[i].y = (v[i].y - mean) * inv;
    v[i].z = (v[i].z - mean) * inv;
    v[i].w = (v[i].w - mean) * inv;
  }
  #pragma unroll
  for (int i = 0; i < 4; ++i) ((float4*)p)[i] = v[i];
}

extern "C" void kernel_launch(void* const* d_in, const int* in_sizes, int n_in,
                              void* d_out, int out_size, void* d_ws, size_t ws_size,
                              hipStream_t stream)
{
  (void)in_sizes; (void)n_in; (void)d_ws; (void)ws_size; (void)out_size;
  const float* x  = (const float*)d_in[0];
  const float* w1 = (const float*)d_in[1];
  const float* b1 = (const float*)d_in[2];
  float* out = (float*)d_out;

  // allow 96KB dynamic LDS (hardware supports 320KB/WG on CDNA5)
  hipFuncSetAttribute(reinterpret_cast<const void*>(ffn_gemm_kernel),
                      hipFuncAttributeMaxDynamicSharedMemorySize, LDS_BYTES);

  // e-major sweep keeps the 256-row x slab and L2-resident w1 hot
  dim3 grid(FDIM / NB, NROW / MB);   // (32, 64)
  ffn_gemm_kernel<<<grid, 256, LDS_BYTES, stream>>>(x, w1, b1, out);
  rownorm_kernel<<<NROW, 256, 0, stream>>>(out);
}
